// InfoNCELoss_28011776705069
// MI455X (gfx1250) — matrix operand order repaired and checked by reading, not compiled, not run on hardware
//
#include <hip/hip_runtime.h>

typedef __attribute__((ext_vector_type(16))) _Float16 v16h;
typedef __attribute__((ext_vector_type(8)))  _Float16 v8h;
typedef __attribute__((ext_vector_type(8)))  float    v8f;

#define D      512
#define MTOT   8192
#define NHALF  4096
#define TINV   10.0f           // 1 / TEMPERATURE
#define NEGBIG (-9.0e16f)      // NEG_INF / TEMPERATURE

// ---------------------------------------------------------------------------
// K1: row-normalize concat(feat1,feat2) -> f16 Fn in workspace; zero nll accum
// ---------------------------------------------------------------------------
__global__ __launch_bounds__(128)
void nce_normalize(const float* __restrict__ f1, const float* __restrict__ f2,
                   _Float16* __restrict__ fn, float* __restrict__ out0) {
  const int row = blockIdx.x;                       // 0..8191
  const float* src = (row < NHALF) ? (f1 + (size_t)row * D)
                                   : (f2 + (size_t)(row - NHALF) * D);
  float ss = 0.f;
  for (int c = threadIdx.x; c < D; c += 128) { float v = src[c]; ss += v * v; }
  #pragma unroll
  for (int off = 16; off > 0; off >>= 1) ss += __shfl_xor(ss, off, 32);
  __shared__ float red[4];
  if ((threadIdx.x & 31) == 0) red[threadIdx.x >> 5] = ss;
  __syncthreads();
  const float tot = red[0] + red[1] + red[2] + red[3];
  const float inv = 1.0f / fmaxf(sqrtf(tot), 1e-8f);   // torch clamps norm at eps
  for (int c = threadIdx.x; c < D; c += 128)
    fn[(size_t)row * D + c] = (_Float16)(src[c] * inv);
  if (row == 0 && threadIdx.x == 0) out0[0] = 0.0f;    // zero nll accumulator
}

// ---------------------------------------------------------------------------
// K2: positive similarities pos[i] = 10 * dot(Fn[i], Fn[i+4096]) (symmetric)
// ---------------------------------------------------------------------------
__global__ __launch_bounds__(64)
void nce_pos(const _Float16* __restrict__ fn, float* __restrict__ pos) {
  const int i = blockIdx.x;                          // 0..4095
  const _Float16* a = fn + (size_t)i * D;
  const _Float16* b = fn + (size_t)(i + NHALF) * D;
  float s = 0.f;
  for (int c = threadIdx.x; c < D; c += 64) s += (float)a[c] * (float)b[c];
  #pragma unroll
  for (int off = 16; off > 0; off >>= 1) s += __shfl_xor(s, off, 32);
  __shared__ float r2[2];
  if ((threadIdx.x & 31) == 0) r2[threadIdx.x >> 5] = s;
  __syncthreads();
  if (threadIdx.x == 0) {
    const float p = (r2[0] + r2[1]) * TINV;
    pos[i] = p;
    pos[i + NHALF] = p;
  }
}

// ---------------------------------------------------------------------------
// K3: fused Fn·Fnᵀ (WMMA f16 -> f32) + online logsumexp + rank counting.
// Block = 128 threads = 4 waves, 2x2 grid of 16x16 WMMA tiles:
// rows [i0, i0+32), sweep all 8192 columns in steps of 32.
// ---------------------------------------------------------------------------
struct RedBuf {
  float m[2][2][16];
  float s[2][2][16];
  int   c[2][2][16];
};

__global__ __launch_bounds__(128)
void nce_gemm_stats(const _Float16* __restrict__ fn, const float* __restrict__ pos,
                    float* __restrict__ out /* [0]=nll, [1..8192]=rank */) {
  __shared__ _Float16 lA[32][D];                          // 32 KB, resident
  __shared__ union { _Float16 B[32][D]; RedBuf red; } uB; // 32 KB, aliased

  const int tid  = threadIdx.x;
  const int lane = tid & 31;
  const int wave = tid >> 5;      // 0..3
  const int wm   = wave >> 1;     // row sub-block 0..1
  const int wn   = wave & 1;      // col sub-block 0..1
  const int half = lane >> 4;     // 0/1 (wave32 half)
  const int ln   = lane & 15;

  const int i0 = blockIdx.x * 32;

  // Load A panel (32 rows x 512 K, f16) once: 2048 x uint4
  {
    const uint4* g = (const uint4*)(fn + (size_t)i0 * D);
    uint4* l = (uint4*)(&lA[0][0]);
    for (int t = tid; t < 32 * D / 8; t += 128) l[t] = g[t];
  }

  // Rows owned by this lane (C layout: lane%16 = N, half selects M or M+8):
  const int rbase = i0 + wm * 16 + half * 8;
  float preg[8], rmax[8], rsum[8];
  int   cnt[8];
  #pragma unroll
  for (int r = 0; r < 8; ++r) {
    preg[r] = pos[rbase + r];
    rmax[r] = -INFINITY;
    rsum[r] = 0.f;
    cnt[r]  = 0;
  }

  const int kAoff = half * 16;   // A frag: lanes16-31 hold K=16..31
  const int kBoff = half * 16;   // B frag: lanes16-31 hold K=16..31
  const int arow  = wm * 16 + ln;
  const int brow  = wn * 16 + ln;

  for (int js = 0; js < MTOT / 32; ++js) {
    const int j0 = js * 32;
    __syncthreads();             // WAR on uB.B
    {
      const uint4* g = (const uint4*)(fn + (size_t)j0 * D);
      uint4* l = (uint4*)(&uB.B[0][0]);
      for (int t = tid; t < 32 * D / 8; t += 128) l[t] = g[t];
      if (js + 1 < MTOT / 32)    // hint next panel toward L2/L0
        __builtin_prefetch(fn + (size_t)(j0 + 32) * D + (size_t)tid * 32, 0, 1);
    }
    __syncthreads();             // RAW on uB.B

    v8f acc = {};
    #pragma unroll
    for (int k0 = 0; k0 < D; k0 += 32) {
      v8h alo = *(const v8h*)&lA[arow][k0 + kAoff];
      v8h ahi = *(const v8h*)&lA[arow][k0 + kAoff + 8];
      v16h a = __builtin_shufflevector(alo, ahi,
                 0,1,2,3,4,5,6,7,8,9,10,11,12,13,14,15);
      v16h b = *(const v16h*)&uB.B[brow][k0 + kBoff];
      acc = __builtin_amdgcn_wmma_f32_16x16x32_f16(false, a, false, b,
                                                   (short)0, acc, false, false);
    }

    // Fold this 16x16 tile into per-row online-softmax + rank state.
    const int colg = j0 + wn * 16 + ln;
    #pragma unroll
    for (int r = 0; r < 8; ++r) {
      const int rowg = rbase + r;
      float v = acc[r] * TINV;
      if (colg == rowg) v = NEGBIG;                    // diag mask (pre-temp in ref)
      const float nm = fmaxf(rmax[r], v);
      rsum[r] = rsum[r] * __expf(rmax[r] - nm) + __expf(v - nm);
      rmax[r] = nm;
      const int pcol = (rowg + NHALF) & (MTOT - 1);
      if (colg != pcol && colg != rowg && v > preg[r]) cnt[r]++;
    }
  }

  // Reduce over the 16 lanes of each half (these lanes hold the same rows,
  // different columns). xor masks 8..1 never cross the half boundary.
  #pragma unroll
  for (int r = 0; r < 8; ++r) {
    float m = rmax[r], s = rsum[r];
    int c = cnt[r];
    #pragma unroll
    for (int off = 8; off >= 1; off >>= 1) {
      const float om = __shfl_xor(m, off, 32);
      const float os = __shfl_xor(s, off, 32);
      const int   oc = __shfl_xor(c, off, 32);
      const float nm = fmaxf(m, om);
      s = s * __expf(m - nm) + os * __expf(om - nm);
      m = nm;
      c += oc;
    }
    rmax[r] = m; rsum[r] = s; cnt[r] = c;
  }

  __syncthreads();               // B panel dead -> safe to alias uB.red
  if (ln == 0) {                 // lanes 0 and 16 of each wave
    #pragma unroll
    for (int r = 0; r < 8; ++r) {
      uB.red.m[wm][wn][half * 8 + r] = rmax[r];
      uB.red.s[wm][wn][half * 8 + r] = rsum[r];
      uB.red.c[wm][wn][half * 8 + r] = cnt[r];
    }
  }
  __syncthreads();
  if (tid < 32) {                // one thread per row of this block
    const int bm = tid >> 4, rr = tid & 15;
    const float m0 = uB.red.m[bm][0][rr], s0 = uB.red.s[bm][0][rr];
    const float m1 = uB.red.m[bm][1][rr], s1 = uB.red.s[bm][1][rr];
    const float nm = fmaxf(m0, m1);
    const float S  = s0 * __expf(m0 - nm) + s1 * __expf(m1 - nm);
    const int   C  = uB.red.c[bm][0][rr] + uB.red.c[bm][1][rr];
    const int rowg = i0 + bm * 16 + rr;
    const float lse = nm + __logf(S);
    atomicAdd(out, (lse - pos[rowg]) * (1.0f / (float)MTOT));
    out[1 + rowg] = (float)C;    // rank of positive
  }
}

// ---------------------------------------------------------------------------
extern "C" void kernel_launch(void* const* d_in, const int* in_sizes, int n_in,
                              void* d_out, int out_size, void* d_ws, size_t ws_size,
                              hipStream_t stream) {
  (void)in_sizes; (void)n_in; (void)out_size; (void)ws_size;
  const float* f1 = (const float*)d_in[0];
  const float* f2 = (const float*)d_in[1];
  float* out = (float*)d_out;

  _Float16* fn  = (_Float16*)d_ws;                               // 8 MB
  float*    pos = (float*)((char*)d_ws +
                           (size_t)MTOT * D * sizeof(_Float16)); // +32 KB

  nce_normalize<<<MTOT, 128, 0, stream>>>(f1, f2, fn, out);
  nce_pos<<<NHALF, 64, 0, stream>>>(fn, pos);
  nce_gemm_stats<<<MTOT / 32, 128, 0, stream>>>(fn, pos, out);
}
